// PandemicGNN_GAT_23021024707457
// MI455X (gfx1250) — compile-verified, matrix operand order checked by reading, not run on hardware
//
#include <hip/hip_runtime.h>
#include <hip/hip_bf16.h>

#define B_    8
#define P_    32
#define N0_   10000
#define KCH   8
#define HIDP_ 4
#define HIDR_ 136
#define BN_   (B_ * N0_)      // 80000
#define E_    160000
#define ET_   (E_ + N0_)      // 170000 edges incl. self loops

typedef __attribute__((ext_vector_type(16))) _Float16 v16h;
typedef __attribute__((ext_vector_type(8)))  _Float16 v8h;
typedef __attribute__((ext_vector_type(8)))  float    v8f;

union V16U { v16h v; v8h h[2]; };

__device__ __forceinline__ unsigned fenc(float f) {
    unsigned u = __float_as_uint(f);
    return (u & 0x80000000u) ? ~u : (u | 0x80000000u);
}
__device__ __forceinline__ float fdec(unsigned e) {
    return (e & 0x80000000u) ? __uint_as_float(e & 0x7FFFFFFFu) : __uint_as_float(~e);
}
__device__ __forceinline__ float lrelu_f(float v, float s) { return v > 0.f ? v : s * v; }

// ---------------------------------------------------------------------------
// Backbone pass A: per-channel sum / sumsq of conv+bias output (BN batch stats)
// ---------------------------------------------------------------------------
__global__ void conv_stats(const float* __restrict__ x, const float* __restrict__ w,
                           const float* __restrict__ bias, float* __restrict__ stats,
                           int ks, int dil, int T) {
    long idx = (long)blockIdx.x * blockDim.x + threadIdx.x;
    long total = (long)B_ * T * N0_;
    float y[KCH];
    if (idx < total) {
        int n = (int)(idx % N0_);
        int t = (int)((idx / N0_) % T);
        int b = (int)(idx / ((long)N0_ * T));
        float xv[32];
        for (int j = 0; j < ks; j++) xv[j] = x[((long)b * P_ + t + j * dil) * N0_ + n];
        for (int k = 0; k < KCH; k++) {
            float a = bias[k];
            for (int j = 0; j < ks; j++) a += w[k * ks + j] * xv[j];
            y[k] = a;
        }
    } else {
        for (int k = 0; k < KCH; k++) y[k] = 0.f;
    }
    for (int k = 0; k < KCH; k++) {
        float s1 = y[k], s2 = y[k] * y[k];
        for (int off = 16; off > 0; off >>= 1) {
            s1 += __shfl_down(s1, off, 32);
            s2 += __shfl_down(s2, off, 32);
        }
        if ((threadIdx.x & 31) == 0) {
            atomicAdd(&stats[2 * k],     s1);
            atomicAdd(&stats[2 * k + 1], s2);
        }
    }
}

// ---------------------------------------------------------------------------
// Backbone pass B: recompute conv, BN affine, adaptive max-pool, tanh -> feat
// ---------------------------------------------------------------------------
__global__ void conv_feat(const float* __restrict__ x, const float* __restrict__ w,
                          const float* __restrict__ bias, const float* __restrict__ gamma,
                          const float* __restrict__ beta, const float* __restrict__ stats,
                          float* __restrict__ feat, int ks, int dil, int T, int pool, int chOff) {
    int idx = blockIdx.x * blockDim.x + threadIdx.x;
    if (idx >= BN_) return;
    int n = idx % N0_, b = idx / N0_;
    float inv_cnt = 1.f / ((float)B_ * (float)T * (float)N0_);
    float xr[P_];
    for (int p = 0; p < P_; p++) xr[p] = x[((long)b * P_ + p) * N0_ + n];
    for (int k = 0; k < KCH; k++) {
        float mu  = stats[2 * k] * inv_cnt;
        float var = stats[2 * k + 1] * inv_cnt - mu * mu;
        float aa  = gamma[k] * rsqrtf(var + 1e-5f);
        float cc  = beta[k] - mu * aa;
        if (pool) {
            for (int i = 0; i < HIDP_; i++) {
                int s0 = (i * T) / HIDP_;
                int e0 = ((i + 1) * T + HIDP_ - 1) / HIDP_;   // ceil
                float m = -3.4e38f;
                for (int t = s0; t < e0; t++) {
                    float yv = bias[k];
                    for (int j = 0; j < ks; j++) yv += w[k * ks + j] * xr[t + j * dil];
                    m = fmaxf(m, aa * yv + cc);
                }
                feat[(long)idx * HIDR_ + chOff + k * HIDP_ + i] = tanhf(m);
            }
        } else {
            float yv = bias[k];
            for (int j = 0; j < ks; j++) yv += w[k * ks + j] * xr[j * dil];
            feat[(long)idx * HIDR_ + chOff + k] = tanhf(aa * yv + cc);
        }
    }
}

// ---------------------------------------------------------------------------
// WMMA GEMM: C[M,N] = act_out( A[M,K] * W[K,N] + bias ), f16 in, f32 acc.
// Concat mode: element(row,k) = k<K1 ? A[row,k] : A2[row,k-K1]; act_in = lrelu.
// Tiles stored in LDS pre-swizzled to WMMA fragment order so every lane loads
// its 16 halves as two contiguous ds_load_b128:
//   perm(k) = ((k&8)<<1) + ((k&16)>>1) + (k&7)   ->  frag(e,g) = row[ g*16 + e ]
// Tile 256x64, 8 waves; each wave: 2 M-frags x 4 N-frags = 8 WMMA per K-chunk.
// ---------------------------------------------------------------------------
#define TM 256
#define TN 64
#define TK 32
#define LDK (TK + 8)   // 40 halves = 80 B row stride (16B-aligned, bank-spread)

__global__ __launch_bounds__(256) void wmma_gemm(
    const float* __restrict__ A, const float* __restrict__ A2,
    const float* __restrict__ W, const float* __restrict__ bias,
    float* __restrict__ C, int M, int N, int K, int K1, int act_in, int act_out) {
    __shared__ _Float16 As[TM][LDK];   // 20.0 KB
    __shared__ _Float16 Bs[TN][LDK];   //  5.0 KB
    const int tid  = threadIdx.x;
    const int wv   = tid >> 5;
    const int lane = tid & 31;
    const int g    = lane >> 4;   // frag K-half select / C M-half select
    const int lr   = lane & 15;
    const int m0   = blockIdx.x * TM;
    const int n0   = blockIdx.y * TN;

    v8f acc[2][4];
#pragma unroll
    for (int m = 0; m < 2; m++)
#pragma unroll
        for (int j = 0; j < 4; j++)
#pragma unroll
            for (int r = 0; r < 8; r++) acc[m][j][r] = 0.f;

    for (int k0 = 0; k0 < K; k0 += TK) {
        // ---- A tile: 8 consecutive K floats -> one 16B ds_store_b128 ----
        // (K and K1 are multiples of 8, so a chunk never straddles A/A2 or K.)
        for (int i = tid; i < TM * (TK / 8); i += 256) {
            int row = i >> 2;
            int q   = i & 3;
            int gr  = m0 + row;
            int gkb = k0 + q * 8;
            v8h hv;
            if (gr < M && gkb + 8 <= K) {
                const float* src = (gkb < K1) ? (A + (long)gr * K1 + gkb)
                                              : (A2 + (long)gr * (K - K1) + (gkb - K1));
                float4 f0 = *(const float4*)(src);
                float4 f1 = *(const float4*)(src + 4);
                float t0 = f0.x, t1 = f0.y, t2 = f0.z, t3 = f0.w;
                float t4 = f1.x, t5 = f1.y, t6 = f1.z, t7 = f1.w;
                if (act_in) {
                    t0 = lrelu_f(t0, 0.01f); t1 = lrelu_f(t1, 0.01f);
                    t2 = lrelu_f(t2, 0.01f); t3 = lrelu_f(t3, 0.01f);
                    t4 = lrelu_f(t4, 0.01f); t5 = lrelu_f(t5, 0.01f);
                    t6 = lrelu_f(t6, 0.01f); t7 = lrelu_f(t7, 0.01f);
                }
                hv[0] = (_Float16)t0; hv[1] = (_Float16)t1;
                hv[2] = (_Float16)t2; hv[3] = (_Float16)t3;
                hv[4] = (_Float16)t4; hv[5] = (_Float16)t5;
                hv[6] = (_Float16)t6; hv[7] = (_Float16)t7;
            } else {
#pragma unroll
                for (int t = 0; t < 8; t++) hv[t] = (_Float16)0.f;
            }
            int pos = ((q & 1) << 4) | ((q & 2) << 2);   // q:0->0, 1->16, 2->8, 3->24
            *(v8h*)&As[row][pos] = hv;
        }
        // ---- B tile: coalesced float4 of 4 cols -> 4 ds_store_b16 (transposed+permuted) ----
        for (int i = tid; i < (TK * TN) / 4; i += 256) {
            int kk  = i >> 4;             // 0..31
            int c4  = (i & 15) << 2;      // 0..60
            int gk  = k0 + kk;
            int gcb = n0 + c4;            // N multiple of 4 -> chunk fully in or out
            float4 f = make_float4(0.f, 0.f, 0.f, 0.f);
            if (gk < K && gcb < N) f = *(const float4*)(W + (long)gk * N + gcb);
            int pos = ((kk & 8) << 1) + ((kk & 16) >> 1) + (kk & 7);
            Bs[c4 + 0][pos] = (_Float16)f.x;
            Bs[c4 + 1][pos] = (_Float16)f.y;
            Bs[c4 + 2][pos] = (_Float16)f.z;
            Bs[c4 + 3][pos] = (_Float16)f.w;
        }
        __syncthreads();

        v16h a[2];
#pragma unroll
        for (int m = 0; m < 2; m++) {
            int row = wv * 32 + m * 16 + lr;
            V16U u;
            u.h[0] = *(const v8h*)&As[row][g * 16];
            u.h[1] = *(const v8h*)&As[row][g * 16 + 8];
            a[m] = u.v;
        }
#pragma unroll
        for (int j = 0; j < 4; j++) {
            int col = j * 16 + lr;
            V16U u;
            u.h[0] = *(const v8h*)&Bs[col][g * 16];
            u.h[1] = *(const v8h*)&Bs[col][g * 16 + 8];
            v16h bfr = u.v;
            acc[0][j] = __builtin_amdgcn_wmma_f32_16x16x32_f16(
                false, a[0], false, bfr, (short)0, acc[0][j], false, false);
            acc[1][j] = __builtin_amdgcn_wmma_f32_16x16x32_f16(
                false, a[1], false, bfr, (short)0, acc[1][j], false, false);
        }
        __syncthreads();
    }

#pragma unroll
    for (int m = 0; m < 2; m++) {
#pragma unroll
        for (int j = 0; j < 4; j++) {
#pragma unroll
            for (int r = 0; r < 8; r++) {
                int row = m0 + wv * 32 + m * 16 + 8 * g + r;  // C: M = r + 8*(lane/16)
                int col = n0 + j * 16 + lr;                   //    N = lane%16
                if (row < M && col < N) {
                    float v = acc[m][j][r];
                    if (bias) v += bias[col];
                    if (act_out) v = lrelu_f(v, 0.01f);
                    C[(long)row * N + col] = v;
                }
            }
        }
    }
}

// ---------------------------------------------------------------------------
// GAT attention pieces
// ---------------------------------------------------------------------------
__global__ void alpha_kernel(const float* __restrict__ h, const float* __restrict__ as,
                             const float* __restrict__ ad, float* __restrict__ asrc,
                             float* __restrict__ adst, int H, int C) {
    int idx = blockIdx.x * blockDim.x + threadIdx.x;
    if (idx >= BN_ * H) return;
    int hh = idx % H, row = idx / H;
    const float* hp = h + (long)row * H * C + (long)hh * C;
    float s = 0.f, d = 0.f;
    for (int c = 0; c < C; c++) { float v = hp[c]; s += v * as[hh * C + c]; d += v * ad[hh * C + c]; }
    asrc[row * H + hh] = s;
    adst[row * H + hh] = d;
}

__global__ void edge_max(const int* __restrict__ ei, const float* __restrict__ asrc,
                         const float* __restrict__ adst, float* __restrict__ albuf,
                         unsigned* __restrict__ amax, int H) {
    int idx = blockIdx.x * blockDim.x + threadIdx.x;
    if (idx >= B_ * ET_) return;
    int e = idx % ET_, b = idx / ET_;
    int s = (e < E_) ? ei[e] : (e - E_);
    int d = (e < E_) ? ei[E_ + e] : (e - E_);
    for (int h = 0; h < H; h++) {
        float al = asrc[(b * N0_ + s) * H + h] + adst[(b * N0_ + d) * H + h];
        al = lrelu_f(al, 0.2f);
        albuf[(long)idx * H + h] = al;
        atomicMax(&amax[(b * N0_ + d) * H + h], fenc(al));
    }
}

__global__ void edge_exp(const int* __restrict__ ei, float* __restrict__ albuf,
                         const unsigned* __restrict__ amax, float* __restrict__ denom, int H) {
    int idx = blockIdx.x * blockDim.x + threadIdx.x;
    if (idx >= B_ * ET_) return;
    int e = idx % ET_, b = idx / ET_;
    int d = (e < E_) ? ei[E_ + e] : (e - E_);
    for (int h = 0; h < H; h++) {
        float al = albuf[(long)idx * H + h];
        float ex = expf(al - fdec(amax[(b * N0_ + d) * H + h]));
        albuf[(long)idx * H + h] = ex;
        atomicAdd(&denom[(b * N0_ + d) * H + h], ex);
    }
}

__global__ void fill_bias(float* __restrict__ out, const float* __restrict__ bias,
                          int total, int HC) {
    int i = blockIdx.x * blockDim.x + threadIdx.x;
    if (i < total) out[i] = bias[i % HC];
}

__global__ void edge_scatter(const int* __restrict__ ei, const float* __restrict__ exbuf,
                             const float* __restrict__ denom, const float* __restrict__ hsrc,
                             float* __restrict__ outp, int H, int C, int total) {
    int idx = blockIdx.x * blockDim.x + threadIdx.x;
    if (idx >= total) return;
    int nc4 = (H * C) >> 2;
    int c4 = idx % nc4;
    int be = idx / nc4;
    int e = be % ET_, b = be / ET_;
    int s = (e < E_) ? ei[e] : (e - E_);
    int d = (e < E_) ? ei[E_ + e] : (e - E_);
    int c = c4 * 4;
    int h = c / C;                       // C multiple of 4: chunk stays in one head
    float ex = exbuf[((long)b * ET_ + e) * H + h];
    float coef = ex / (denom[(b * N0_ + d) * H + h] + 1e-16f);
    const float4 hv = *(const float4*)(hsrc + (long)(b * N0_ + s) * (H * C) + c);
    float* op = outp + (long)(b * N0_ + d) * (long)(H * C) + c;
    atomicAdd(op + 0, coef * hv.x);
    atomicAdd(op + 1, coef * hv.y);
    atomicAdd(op + 2, coef * hv.z);
    atomicAdd(op + 3, coef * hv.w);
}

// ---------------------------------------------------------------------------
// Output head: out[row] = res[row,:] . Wo2 + bo2
// ---------------------------------------------------------------------------
__global__ void out_matvec(const float* __restrict__ res, const float* __restrict__ w,
                           const float* __restrict__ b2, float* __restrict__ out, int M, int K) {
    int row = blockIdx.x * blockDim.x + threadIdx.x;
    if (row >= M) return;
    const float* r = res + (long)row * K;
    float s = b2[0];
    for (int c = 0; c < K; c++) s += r[c] * w[c];
    out[row] = s;
}

// ---------------------------------------------------------------------------
// Host launch: full pipeline on `stream` (graph-capture safe)
// ---------------------------------------------------------------------------
extern "C" void kernel_launch(void* const* d_in, const int* in_sizes, int n_in,
                              void* d_out, int out_size, void* d_ws, size_t ws_size,
                              hipStream_t stream) {
    const float* x   = (const float*)d_in[0];
    const float* W1  = (const float*)d_in[21];
    const float* as1 = (const float*)d_in[22];
    const float* ad1 = (const float*)d_in[23];
    const float* bb1 = (const float*)d_in[24];
    const float* W2  = (const float*)d_in[25];
    const float* as2 = (const float*)d_in[26];
    const float* ad2 = (const float*)d_in[27];
    const float* bb2 = (const float*)d_in[28];
    const float* Wo  = (const float*)d_in[29];
    const float* bo  = (const float*)d_in[30];
    const float* Wo2 = (const float*)d_in[31];
    const float* bo2 = (const float*)d_in[32];
    const int*   ei  = (const int*)d_in[33];

    // workspace layout (peak ~231 MB)
    char* ws = (char*)d_ws;
    const size_t SZ_FEAT = (size_t)BN_ * HIDR_ * 4;   // 43.52 MB
    const size_t SZ_BUF  = (size_t)BN_ * 272 * 4;     // 87.04 MB
    const size_t SZ_EDGE = (size_t)B_ * ET_ * 2 * 4;  // 10.88 MB
    const size_t SZ_AL   = (size_t)BN_ * 2 * 4;       // 0.64 MB
    const size_t O_FEAT = 0;
    const size_t O_BUFA = O_FEAT + SZ_FEAT;           // h1 -> h2 -> res
    const size_t O_BUFB = O_BUFA + SZ_BUF;            // g1 -> g2
    const size_t O_EDGE = O_BUFB + SZ_BUF;
    const size_t O_ASRC = O_EDGE + SZ_EDGE;
    const size_t O_ADST = O_ASRC + SZ_AL;
    const size_t O_AMAX = O_ADST + SZ_AL;
    const size_t O_DEN  = O_AMAX + SZ_AL;
    const size_t O_STAT = O_DEN + SZ_AL;

    float*    feat  = (float*)(ws + O_FEAT);
    float*    bufA  = (float*)(ws + O_BUFA);
    float*    bufB  = (float*)(ws + O_BUFB);
    float*    albuf = (float*)(ws + O_EDGE);
    float*    asrc  = (float*)(ws + O_ASRC);
    float*    adst  = (float*)(ws + O_ADST);
    unsigned* amax  = (unsigned*)(ws + O_AMAX);
    float*    denom = (float*)(ws + O_DEN);
    float*    stats = (float*)(ws + O_STAT);

    dim3 blk(256);

    // ---- backbone ----
    struct Br { int ks, dil, T, pool, off; };
    const Br br[5] = { {3, 1, 30, 1, 0}, {5, 1, 28, 1, 32}, {3, 2, 28, 1, 64},
                       {5, 2, 24, 1, 96}, {P_, 1, 1, 0, 128} };
    hipMemsetAsync(stats, 0, 5 * 2 * KCH * sizeof(float), stream);
    for (int i = 0; i < 5; i++) {
        const float* wb  = (const float*)d_in[1 + 4 * i];
        const float* bb  = (const float*)d_in[2 + 4 * i];
        int total = B_ * br[i].T * N0_;
        conv_stats<<<(total + 255) / 256, blk, 0, stream>>>(
            x, wb, bb, stats + i * 2 * KCH, br[i].ks, br[i].dil, br[i].T);
    }
    for (int i = 0; i < 5; i++) {
        const float* wb  = (const float*)d_in[1 + 4 * i];
        const float* bb  = (const float*)d_in[2 + 4 * i];
        const float* gb  = (const float*)d_in[3 + 4 * i];
        const float* beb = (const float*)d_in[4 + 4 * i];
        conv_feat<<<(BN_ + 255) / 256, blk, 0, stream>>>(
            x, wb, bb, gb, beb, stats + i * 2 * KCH, feat,
            br[i].ks, br[i].dil, br[i].T, br[i].pool, br[i].off);
    }

    const int etot = B_ * ET_;

    // ---- GAT layer 1 (H=2, C=136, HC=272) ----
    wmma_gemm<<<dim3((BN_ + TM - 1) / TM, (272 + TN - 1) / TN), blk, 0, stream>>>(
        feat, feat, W1, nullptr, bufA, BN_, 272, HIDR_, HIDR_, 0, 0);          // h1
    alpha_kernel<<<(BN_ * 2 + 255) / 256, blk, 0, stream>>>(bufA, as1, ad1, asrc, adst, 2, 136);
    hipMemsetAsync(amax, 0, SZ_AL, stream);    // enc(0) acts as -inf
    hipMemsetAsync(denom, 0, SZ_AL, stream);
    edge_max<<<(etot + 255) / 256, blk, 0, stream>>>(ei, asrc, adst, albuf, amax, 2);
    edge_exp<<<(etot + 255) / 256, blk, 0, stream>>>(ei, albuf, amax, denom, 2);
    fill_bias<<<(BN_ * 272 + 255) / 256, blk, 0, stream>>>(bufB, bb1, BN_ * 272, 272);
    {
        int tot = etot * (272 / 4);
        edge_scatter<<<(tot + 255) / 256, blk, 0, stream>>>(ei, albuf, denom, bufA, bufB, 2, 136, tot);
    }                                                                           // g1 in bufB

    // ---- GAT layer 2 (H=1, C=136) ----
    wmma_gemm<<<dim3((BN_ + TM - 1) / TM, (136 + TN - 1) / TN), blk, 0, stream>>>(
        bufB, bufB, W2, nullptr, bufA, BN_, 136, 272, 272, 0, 0);              // h2
    alpha_kernel<<<(BN_ + 255) / 256, blk, 0, stream>>>(bufA, as2, ad2, asrc, adst, 1, 136);
    hipMemsetAsync(amax, 0, (size_t)BN_ * 4, stream);
    hipMemsetAsync(denom, 0, (size_t)BN_ * 4, stream);
    edge_max<<<(etot + 255) / 256, blk, 0, stream>>>(ei, asrc, adst, albuf, amax, 1);
    edge_exp<<<(etot + 255) / 256, blk, 0, stream>>>(ei, albuf, amax, denom, 1);
    fill_bias<<<(BN_ * 136 + 255) / 256, blk, 0, stream>>>(bufB, bb2, BN_ * 136, 136);
    {
        int tot = etot * (136 / 4);
        edge_scatter<<<(tot + 255) / 256, blk, 0, stream>>>(ei, albuf, denom, bufA, bufB, 1, 136, tot);
    }                                                                           // g2 in bufB

    // ---- head: res = lrelu( lrelu(concat[g2, feat]) @ Wo + bo ) ; out = res @ Wo2 + bo2 ----
    wmma_gemm<<<dim3((BN_ + TM - 1) / TM, (136 + TN - 1) / TN), blk, 0, stream>>>(
        bufB, feat, Wo, bo, bufA, BN_, 136, 272, 136, 1, 1);                   // res
    out_matvec<<<(BN_ + 255) / 256, blk, 0, stream>>>(bufA, Wo2, bo2, (float*)d_out, BN_, 136);
}
